// QuestionSpecificMLP_17179869184423
// MI455X (gfx1250) — matrix operand order repaired and checked by reading, not compiled, not run on hardware
//
#include <hip/hip_runtime.h>
#include <hip/hip_bf16.h>
#include <cstddef>

// ---------------------------------------------------------------------------
// QuestionSpecificMLP — CDNA5 (gfx1250) WMMA implementation, round 3.
//
// - Trunk GEMMs: v_wmma_f32_16x16x32_f16, f32 accumulate, f16 activations.
// - Layer shapes are template parameters -> constant K-trip counts, immediate
//   -offset epilogue stores (kills the v_add_nc_u64 address chains), and the
//   K-tail clamp folds away for layers with KW == KPAD.
// - Double-buffered LDS staging: one barrier per K-step; next tile's global
//   loads overlap current tile's WMMAs.
// - x is pre-converted to zero-padded f16 [B,416] so all A staging is
//   unguarded 16B-aligned v8h loads (fallback path if d_ws is too small).
// - Head: per-row gathered 192x16 matvec (never computes dense 2048 logits).
// ---------------------------------------------------------------------------

typedef __attribute__((ext_vector_type(16))) _Float16 v16h;
typedef __attribute__((ext_vector_type(8)))  _Float16 v8h;
typedef __attribute__((ext_vector_type(8)))  float    v8f;

union V16 { v16h v; v8h h[2]; };

#define BM 128
#define BN 64
#define BK 32
#define LDS_STRIDE 40   // halves; 80 B/row -> every 8-half offset is 16B aligned

// C[M,N] = relu(A[M,KPAD] @ W[KW,N] + bias), C is f16.
template <bool A_HALF, int N, int KPAD, int KW, int LDA>
__global__ __launch_bounds__(256)
void gemm_relu_f16(const void* __restrict__ Aptr,
                   const float* __restrict__ W,
                   const float* __restrict__ bias,
                   _Float16* __restrict__ Cout)
{
    __shared__ _Float16 As[2][BM][LDS_STRIDE];
    __shared__ _Float16 Bs[2][BN][LDS_STRIDE];   // Bs[b][n][k], N-major

    const int t       = threadIdx.x;
    const int lane    = t & 31;
    const int wid     = t >> 5;           // 0..7
    const int half    = lane >> 4;        // 0/1
    const int lmod    = lane & 15;
    const int wm      = wid & 3;          // 4 wave-rows  (32 rows each)
    const int wn      = wid >> 2;         // 2 wave-cols  (32 cols each)
    const int rowBase = wm * 32;
    const int colBase = wn * 32;

    const int blockN = blockIdx.x * BN;
    const int blockM = blockIdx.y * BM;

    // global->LDS staging assignments
    const int arow  = t >> 1;             // 0..127
    const int acol0 = (t & 1) << 4;       // 0 or 16
    const int bl    = t << 3;             // 0..2040
    const int bkr   = bl >> 6;            // 0..31 (k within tile)
    const int bnc   = bl & 63;            // 0..56 step 8 (n within tile)

    // ---- staging helpers ----
    auto stageA = [&](int kt, int buf) {
        const size_t gm = (size_t)blockM + arow;
        const int k0 = kt * BK;
        if (A_HALF) {
            const _Float16* Ag = (const _Float16*)Aptr + gm * (size_t)LDA + k0 + acol0;
            *(v8h*)&As[buf][arow][acol0]     = *(const v8h*)Ag;
            *(v8h*)&As[buf][arow][acol0 + 8] = *(const v8h*)(Ag + 8);
        } else {
            const float* Ag = (const float*)Aptr + gm * (size_t)LDA;
            #pragma unroll
            for (int i = 0; i < 16; ++i) {
                const int gk = k0 + acol0 + i;
                const int ck = (gk < KW) ? gk : (KW - 1);
                const float v = Ag[ck];
                As[buf][arow][acol0 + i] = (_Float16)((gk < KW) ? v : 0.0f);
            }
        }
    };
    auto stageB = [&](int kt, int buf) {
        const int gk = kt * BK + bkr;
        if (KW == KPAD) {
            const float* Wg = W + (size_t)gk * N + blockN + bnc;
            const float4 f0 = *(const float4*)Wg;
            const float4 f1 = *(const float4*)(Wg + 4);
            Bs[buf][bnc + 0][bkr] = (_Float16)f0.x;
            Bs[buf][bnc + 1][bkr] = (_Float16)f0.y;
            Bs[buf][bnc + 2][bkr] = (_Float16)f0.z;
            Bs[buf][bnc + 3][bkr] = (_Float16)f0.w;
            Bs[buf][bnc + 4][bkr] = (_Float16)f1.x;
            Bs[buf][bnc + 5][bkr] = (_Float16)f1.y;
            Bs[buf][bnc + 6][bkr] = (_Float16)f1.z;
            Bs[buf][bnc + 7][bkr] = (_Float16)f1.w;
        } else {
            const int  ck = (gk < KW) ? gk : (KW - 1);
            const bool ok = (gk < KW);
            const float* Wg = W + (size_t)ck * N + blockN + bnc;
            const float4 f0 = *(const float4*)Wg;
            const float4 f1 = *(const float4*)(Wg + 4);
            Bs[buf][bnc + 0][bkr] = (_Float16)(ok ? f0.x : 0.0f);
            Bs[buf][bnc + 1][bkr] = (_Float16)(ok ? f0.y : 0.0f);
            Bs[buf][bnc + 2][bkr] = (_Float16)(ok ? f0.z : 0.0f);
            Bs[buf][bnc + 3][bkr] = (_Float16)(ok ? f0.w : 0.0f);
            Bs[buf][bnc + 4][bkr] = (_Float16)(ok ? f1.x : 0.0f);
            Bs[buf][bnc + 5][bkr] = (_Float16)(ok ? f1.y : 0.0f);
            Bs[buf][bnc + 6][bkr] = (_Float16)(ok ? f1.z : 0.0f);
            Bs[buf][bnc + 7][bkr] = (_Float16)(ok ? f1.w : 0.0f);
        }
    };

    v8f acc[2][2];
    #pragma unroll
    for (int i = 0; i < 2; ++i)
        #pragma unroll
        for (int j = 0; j < 2; ++j)
            acc[i][j] = (v8f)0.0f;

    constexpr int kTiles = KPAD / BK;

    // prologue: stage tile 0
    stageA(0, 0);
    stageB(0, 0);
    __syncthreads();

    int cur = 0;
    for (int kt = 0; kt < kTiles; ++kt) {
        const int nxt = cur ^ 1;

        // ---- fragments from current buffer ----
        // A 16x32 f16: lanes 0-15 (M=lmod): VGPR0-3=K0..7, VGPR4-7=K16..23
        //              lanes 16-31         : VGPR0-3=K8..15, VGPR4-7=K24..31
        V16 afrag[2];
        #pragma unroll
        for (int i = 0; i < 2; ++i) {
            const int m  = rowBase + i * 16 + lmod;
            const int kO = half * 8;
            afrag[i].h[0] = *(const v8h*)&As[cur][m][kO];
            afrag[i].h[1] = *(const v8h*)&As[cur][m][kO + 16];
        }
        // B 32x16 f16: lanes 0-15 (N=lmod): K0..7,K16..23; lanes 16-31: K8..15,K24..31
        V16 bfrag[2];
        #pragma unroll
        for (int j = 0; j < 2; ++j) {
            const int n  = colBase + j * 16 + lmod;
            const int kO = half * 8;
            bfrag[j].h[0] = *(const v8h*)&Bs[cur][n][kO];
            bfrag[j].h[1] = *(const v8h*)&Bs[cur][n][kO + 16];
        }

        // ---- stage next tile into the other buffer (overlaps WMMA) ----
        if (kt + 1 < kTiles) {
            stageA(kt + 1, nxt);
            stageB(kt + 1, nxt);
        }

        // ---- WMMA ----
        #pragma unroll
        for (int i = 0; i < 2; ++i)
            #pragma unroll
            for (int j = 0; j < 2; ++j)
                acc[i][j] = __builtin_amdgcn_wmma_f32_16x16x32_f16(
                    false, afrag[i].v, false, bfrag[j].v,
                    (short)0, acc[i][j], false, false);

        __syncthreads();   // next buffer written; current buffer free for reuse
        cur = nxt;
    }

    // ---- epilogue: bias + ReLU, store f16 (N constexpr -> immediate offsets) ----
    // D layout: lane l, VGPR r -> m = r + 8*(l>=16), n = l%16
    #pragma unroll
    for (int i = 0; i < 2; ++i) {
        #pragma unroll
        for (int j = 0; j < 2; ++j) {
            const int gn     = blockN + colBase + j * 16 + lmod;
            const int gmBase = blockM + rowBase + i * 16 + 8 * half;
            const float bv   = bias[gn];
            _Float16* outp   = Cout + (size_t)gmBase * N + gn;
            #pragma unroll
            for (int r = 0; r < 8; ++r) {
                float v = acc[i][j][r] + bv;
                v = v > 0.0f ? v : 0.0f;
                outp[(size_t)r * N] = (_Float16)v;
            }
        }
    }
}

// ---------------------------------------------------------------------------
// x [B,385] f32 -> xh [B,416] f16, zero-padded (branchless clamp+select).
// ---------------------------------------------------------------------------
__global__ __launch_bounds__(256)
void convert_pad_f16(const float* __restrict__ x, _Float16* __restrict__ xh)
{
    const size_t idx = (size_t)blockIdx.x * 256 + threadIdx.x;   // over B*416
    const int row = (int)(idx / 416);
    const int k   = (int)(idx % 416);
    const int ck  = (k < 385) ? k : 384;
    const float v = x[(size_t)row * 385 + ck];
    xh[idx] = (_Float16)((k < 385) ? v : 0.0f);
}

// ---------------------------------------------------------------------------
// Head: per-row gathered 192x16 matvec.
// out[b, j] = bh[base+j] + sum_k h3[b,k] * Wh[k, base+j],
// base = (qid[b]*2 + isc[b]) * 16.  16 rows per block, 16 threads per row.
// ---------------------------------------------------------------------------
__global__ __launch_bounds__(256)
void head_gather(const _Float16* __restrict__ h3,
                 const float* __restrict__ Wh,
                 const float* __restrict__ bh,
                 const int* __restrict__ qid,
                 const int* __restrict__ isc,
                 float* __restrict__ out)
{
    __shared__ float sh[16][193];   // 193 stride -> conflict-free broadcast

    const int t   = threadIdx.x;
    const int lr  = t >> 4;         // local row 0..15
    const int j   = t & 15;         // head column 0..15
    const int row = blockIdx.x * 16 + lr;

    {
        const _Float16* hrow = h3 + (size_t)row * 192;
        #pragma unroll
        for (int i = j; i < 192; i += 16)
            sh[lr][i] = (float)hrow[i];
    }
    __syncthreads();

    const int base = (qid[row] * 2 + isc[row]) * 16;
    const float* __restrict__ Wcol = Wh + base + j;   // stride 2048 over k

    float acc = bh[base + j];
    #pragma unroll 8
    for (int k = 0; k < 192; ++k)
        acc = fmaf(sh[lr][k], Wcol[(size_t)k * 2048], acc);

    out[(size_t)row * 16 + j] = acc;
}

// ---------------------------------------------------------------------------
// Launch
// ---------------------------------------------------------------------------
extern "C" void kernel_launch(void* const* d_in, const int* in_sizes, int n_in,
                              void* d_out, int out_size, void* d_ws, size_t ws_size,
                              hipStream_t stream) {
    (void)in_sizes; (void)n_in; (void)out_size;

    const float* x   = (const float*)d_in[0];
    const float* W1  = (const float*)d_in[1];
    const float* b1  = (const float*)d_in[2];
    const float* W2  = (const float*)d_in[3];
    const float* b2  = (const float*)d_in[4];
    const float* W3  = (const float*)d_in[5];
    const float* b3  = (const float*)d_in[6];
    const float* Wh  = (const float*)d_in[7];
    const float* bh  = (const float*)d_in[8];
    const int*   qid = (const int*)d_in[9];
    const int*   isc = (const int*)d_in[10];
    float*       out = (float*)d_out;

    const int    B  = 65536;
    const size_t KP = 416;   // 385 padded to 13*32

    const size_t bytes_xh = (size_t)B * KP  * sizeof(_Float16);  // 54.5 MB
    const size_t bytes_h1 = (size_t)B * 512 * sizeof(_Float16);  // 64 MB
    const size_t bytes_h2 = (size_t)B * 256 * sizeof(_Float16);  // 32 MB
    const size_t bytes_h3 = (size_t)B * 192 * sizeof(_Float16);  // 24 MB

    dim3 blk(256);
    char* p = (char*)d_ws;

    _Float16* h1;
    if (ws_size >= bytes_xh + bytes_h1 + bytes_h2 + bytes_h3) {
        // preferred: pre-convert x to padded f16, fully vectorized layer 1
        _Float16* xh = (_Float16*)p;  p += bytes_xh;
        h1 = (_Float16*)p;

        const size_t nConv = (size_t)B * KP;
        convert_pad_f16<<<dim3((unsigned)(nConv / 256)), blk, 0, stream>>>(x, xh);

        gemm_relu_f16<true, 512, 416, 385, 416>
            <<<dim3(512 / BN, B / BM), blk, 0, stream>>>(
                (const void*)xh, W1, b1, h1);
    } else {
        // fallback: consume f32 x directly (branchless clamped loads)
        h1 = (_Float16*)p;
        gemm_relu_f16<false, 512, 416, 385, 385>
            <<<dim3(512 / BN, B / BM), blk, 0, stream>>>(
                (const void*)x, W1, b1, h1);
    }

    _Float16* h2 = h1 + (size_t)B * 512;
    _Float16* h3 = h2 + (size_t)B * 256;

    gemm_relu_f16<true, 256, 512, 512, 512>
        <<<dim3(256 / BN, B / BM), blk, 0, stream>>>(
            (const void*)h1, W2, b2, h2);

    gemm_relu_f16<true, 192, 256, 256, 256>
        <<<dim3(192 / BN, B / BM), blk, 0, stream>>>(
            (const void*)h2, W3, b3, h3);

    head_gather<<<dim3(B / 16), blk, 0, stream>>>(h3, Wh, bh, qid, isc, out);
}
